// CausalMultiHeadSelfAttention_1494648619247
// MI455X (gfx1250) — matrix operand order repaired and checked by reading, not compiled, not run on hardware
//
#include <hip/hip_runtime.h>

#if __has_include(<hip/amd_detail/amd_gfx1250_TDM.h>)
#define TDM_ARGS6 1
#endif

typedef __attribute__((ext_vector_type(16))) _Float16     v16h;
typedef __attribute__((ext_vector_type(8)))  float        v8f;
typedef __attribute__((ext_vector_type(4)))  unsigned int v4u;
typedef __attribute__((ext_vector_type(4)))  int          v4i;
typedef __attribute__((ext_vector_type(8)))  int          v8i;

#define BDIM 2
#define TDIM 2048
#define DDIM 1024
#define HDIM 16
#define HD   64
#define MTOT (BDIM * TDIM)
#define NEG_BIG (-1.0e30f)

union frag16 {
    v16h h;
    uint4 q[2];
};

static __device__ __forceinline__ v8f wmma_f16(const v16h& a, const v16h& b, const v8f& c) {
    return __builtin_amdgcn_wmma_f32_16x16x32_f16(false, a, false, b, (short)0, c, false, false);
}

static __device__ __forceinline__ unsigned lds_off(const void* p) {
    return (unsigned)(size_t)p;  // flat LDS-aperture address: low 32 bits = LDS byte offset
}

// ---------------------------------------------------------------------------
// TDM: async 2D f16 tile load Global -> LDS with hardware row padding.
//   tile_d0 : contiguous elements per row, tile_d1 : rows
//   stride0 : elements between consecutive rows in memory
//   pad codes: interval 3 => every 16 DW, 4 => every 32 DW; amount 3 => 4 DW
// ---------------------------------------------------------------------------
static __device__ __forceinline__ void tdm_load_2d_f16(const void* gptr, unsigned lds_addr,
                                                       unsigned tile_d0, unsigned tile_d1,
                                                       unsigned long long stride0,
                                                       unsigned pad_int, unsigned pad_amt) {
    unsigned long long ga = (unsigned long long)(size_t)gptr;
    v4u g0;
    g0[0] = 1u;                                               // count=1 (valid), user mode
    g0[1] = lds_addr;                                         // LDS byte address
    g0[2] = (unsigned)(ga & 0xffffffffull);                   // global_addr[31:0]
    g0[3] = (unsigned)((ga >> 32) & 0x01ffffffull) | (2u << 30);  // global_addr[56:32] | type=2
    v8i g1;
    g1[0] = (int)((1u << 16)            // data_size = 2 bytes
                | (1u << 20)            // pad_enable
                | (pad_int << 22)       // pad_interval code
                | (pad_amt << 25));     // pad_amount code
    g1[1] = (int)((tile_d0 & 0xffffu) << 16);                         // tensor_dim0[15:0]
    g1[2] = (int)(((tile_d0 >> 16) & 0xffffu) | ((tile_d1 & 0xffffu) << 16));  // td0 hi | td1 lo
    g1[3] = (int)(((tile_d1 >> 16) & 0xffffu) | ((tile_d0 & 0xffffu) << 16));  // td1 hi | tile_dim0
    g1[4] = (int)(tile_d1 & 0xffffu);                                 // tile_dim1 | tile_dim2=0
    g1[5] = (int)(unsigned)(stride0 & 0xffffffffull);                 // tensor_dim0_stride lo
    g1[6] = (int)(unsigned)((stride0 >> 32) & 0xffffull);             // stride hi | dim1_stride lo=0
    g1[7] = 0;
    v4i z4 = {0, 0, 0, 0};
#ifdef TDM_ARGS6
    v8i z8 = {0, 0, 0, 0, 0, 0, 0, 0};
    __builtin_amdgcn_tensor_load_to_lds(g0, g1, z4, z4, z8, 0);
#else
    __builtin_amdgcn_tensor_load_to_lds(g0, g1, z4, z4, 0);
#endif
}

// ---------------------------------------------------------------------------
// elementwise fp32 -> f16
// ---------------------------------------------------------------------------
__global__ void f32_to_f16_kernel(const float* __restrict__ in, _Float16* __restrict__ out, int n) {
    int i = blockIdx.x * blockDim.x + threadIdx.x;
    if (i < n) out[i] = (_Float16)in[i];
}

// ---------------------------------------------------------------------------
// W [N,K] fp32 -> fragment-friendly f16 layout: out[((k>>5)*N + n)*32 + (k&31)]
// ---------------------------------------------------------------------------
__global__ void swizzle_w_kernel(const float* __restrict__ W, _Float16* __restrict__ out,
                                 int N, int K) {
    int i = blockIdx.x * blockDim.x + threadIdx.x;
    if (i < N * K) {
        int kp = i / (N * 32);
        int r  = i - kp * (N * 32);
        int n  = r >> 5;
        int kk = r & 31;
        out[i] = (_Float16)W[(size_t)n * K + kp * 32 + kk];
    }
}

// ---------------------------------------------------------------------------
// WMMA GEMM:  out = (A[M,K] * W^T + bias) * scale
// A row-major f16; Bsw = swizzled weight [K/32][N][32] f16; bias fp32.
// mode 0: f16 head-split [B,H,T,64]; mode 1: f16 head-split transposed
// [B,H,64,T]; mode 2: fp32 row-major [M,N].
// block = 256 (8 waves); tile 128(M) x 64(N); K step 32.
// TDM double-buffered staging: wave 0 issues tensor_load_to_lds for A & B
// tiles; TENSORcnt + block barrier publish each tile.
// ---------------------------------------------------------------------------
__global__ __launch_bounds__(256) void gemm_f16_kernel(const _Float16* __restrict__ A,
                                                       const _Float16* __restrict__ Bsw,
                                                       const float* __restrict__ bias,
                                                       void* __restrict__ outp,
                                                       int M, int N, int K,
                                                       int mode, float scale) {
    __shared__ _Float16 As[2][128 * 40];  // [m][k'], row = 16 DW + 4 DW pad (stride 40 halves)
    __shared__ _Float16 Bs[2][64 * 40];   // [n][k'], same padding

    const int tid  = threadIdx.x;
    const int lane = tid & 31;
    const int wave = tid >> 5;
    const int wm   = (wave & 3) * 32;
    const int wn   = (wave >> 2) * 32;
    const int half = lane >> 4;
    const int col  = lane & 15;

    const int m0 = blockIdx.y * 128;
    const int n0 = blockIdx.x * 64;
    const int nk = K / 32;

    if (wave == 0) {
        tdm_load_2d_f16(A + (size_t)m0 * K,            lds_off(&As[0][0]), 32, 128, (unsigned long long)K, 3, 3);
        tdm_load_2d_f16(Bsw + (size_t)n0 * 32,         lds_off(&Bs[0][0]), 32, 64,  32ull,                 3, 3);
    }

    v8f acc[2][2] = {};

    for (int kt = 0; kt < nk; kt++) {
        if (wave == 0) __builtin_amdgcn_s_wait_tensorcnt((short)0);
        __syncthreads();  // tile kt ready; all reads of the other buffer finished
        if ((kt + 1) < nk && wave == 0) {
            const int nxt = (kt + 1) & 1;
            tdm_load_2d_f16(A + (size_t)m0 * K + (kt + 1) * 32,        lds_off(&As[nxt][0]), 32, 128, (unsigned long long)K, 3, 3);
            tdm_load_2d_f16(Bsw + ((size_t)(kt + 1) * N + n0) * 32,    lds_off(&Bs[nxt][0]), 32, 64,  32ull,                 3, 3);
        }

        const int cur = kt & 1;
        frag16 af[2], bf[2];
        for (int mi = 0; mi < 2; mi++) {
            const uint4* p = (const uint4*)(&As[cur][(wm + mi * 16 + col) * 40]);
            af[mi].q[0] = p[half];        // k' = half*8 + j
            af[mi].q[1] = p[2 + half];    // k' = 16 + half*8 + j
        }
        for (int ni = 0; ni < 2; ni++) {
            const uint4* p = (const uint4*)(&Bs[cur][(wn + ni * 16 + col) * 40]);
            bf[ni].q[0] = p[half];          // k' = half*8 + j
            bf[ni].q[1] = p[2 + half];      // k' = 16 + half*8 + j
        }
        for (int mi = 0; mi < 2; mi++)
            for (int ni = 0; ni < 2; ni++)
                acc[mi][ni] = wmma_f16(af[mi].h, bf[ni].h, acc[mi][ni]);
    }

    // epilogue
    for (int mi = 0; mi < 2; mi++) {
        for (int ni = 0; ni < 2; ni++) {
            for (int i = 0; i < 8; i++) {
                int gm = m0 + wm + mi * 16 + i + 8 * half;
                int gn = n0 + wn + ni * 16 + col;
                float val = (acc[mi][ni][i] + bias[gn]) * scale;
                if (mode == 2) {
                    ((float*)outp)[(size_t)gm * N + gn] = val;
                } else {
                    _Float16 hv = (_Float16)val;
                    int bb = gm >> 11, tt = gm & (TDIM - 1);
                    int hh = gn >> 6,  dd = gn & (HD - 1);
                    _Float16* o = (_Float16*)outp;
                    if (mode == 0)
                        o[(((size_t)bb * HDIM + hh) * TDIM + tt) * HD + dd] = hv;
                    else
                        o[(((size_t)bb * HDIM + hh) * HD + dd) * TDIM + tt] = hv;
                }
            }
        }
    }
}

// ---------------------------------------------------------------------------
// Flash attention.
// Q [B,H,T,64] f16 (pre-scaled); K [B,H,T,64] f16; Vt [B,H,64,T] f16.
// ctx [B,T,D] f16.  grid = (T/128, B*H); block = 256 (8 waves x 16 q rows).
// K/V tiles (32 keys) staged by TDM, double-buffered.
// ---------------------------------------------------------------------------
__global__ __launch_bounds__(256) void attn_kernel(const _Float16* __restrict__ Q,
                                                   const _Float16* __restrict__ Kg,
                                                   const _Float16* __restrict__ Vt,
                                                   _Float16* __restrict__ ctx) {
    __shared__ _Float16 Ks[2][32 * 72];   // [key][d], row = 32 DW + 4 DW pad (stride 72 halves)
    __shared__ _Float16 Vs[2][64 * 40];   // [d][key], row = 16 DW + 4 DW pad (stride 40 halves)
    __shared__ _Float16 Ps[8 * 16 * 40];  // per-wave P: 16 rows x 32 keys (+pad)

    const int tid  = threadIdx.x;
    const int lane = tid & 31;
    const int wave = tid >> 5;
    const int half = lane >> 4;
    const int col  = lane & 15;

    const int bh = blockIdx.y;         // b*16 + h
    const int q0 = blockIdx.x * 128;
    const int qwave = q0 + wave * 16;

    // Q fragments (d 0..31, 32..63) stay in registers
    frag16 qf[2];
    {
        const uint4* p = (const uint4*)(Q + ((size_t)bh * TDIM + qwave + col) * HD);
        for (int f = 0; f < 2; f++) {
            qf[f].q[0] = p[f * 4 + half];
            qf[f].q[1] = p[f * 4 + 2 + half];
        }
    }

    v8f acc[4] = {};
    float mrun[8], lrun[8];
    for (int i = 0; i < 8; i++) { mrun[i] = NEG_BIG; lrun[i] = 0.0f; }

    const int ntiles = ((q0 + 127) >> 5) + 1;

    if (wave == 0) {
        tdm_load_2d_f16(Kg + (size_t)bh * TDIM * HD, lds_off(&Ks[0][0]), 64, 32, (unsigned long long)HD,   4, 3);
        tdm_load_2d_f16(Vt + (size_t)bh * HD * TDIM, lds_off(&Vs[0][0]), 32, 64, (unsigned long long)TDIM, 3, 3);
    }

    for (int it = 0; it < ntiles; it++) {
        const int kt0 = it * 32;
        if (wave == 0) __builtin_amdgcn_s_wait_tensorcnt((short)0);
        __syncthreads();  // tile `it` ready; other buffer free for reuse
        if ((it + 1) < ntiles && wave == 0) {
            const int nxt = (it + 1) & 1;
            const int kn = (it + 1) * 32;
            tdm_load_2d_f16(Kg + ((size_t)bh * TDIM + kn) * HD, lds_off(&Ks[nxt][0]), 64, 32, (unsigned long long)HD,   4, 3);
            tdm_load_2d_f16(Vt + (size_t)bh * HD * TDIM + kn,   lds_off(&Vs[nxt][0]), 32, 64, (unsigned long long)TDIM, 3, 3);
        }
        const int cur = it & 1;

        if (kt0 <= qwave + 15) {  // wave-uniform: skip fully-masked key tiles
            // scores: two 16x16 tiles (keys 0-15, 16-31)
            v8f st[2];
            for (int kt = 0; kt < 2; kt++) {
                v8f sc = {};
                for (int d0 = 0; d0 < 2; d0++) {
                    frag16 kb;
                    const uint4* p = (const uint4*)(&Ks[cur][(kt * 16 + col) * 72]);
                    kb.q[0] = p[d0 * 4 + half];          // d = d0*32 + half*8 + 0..7
                    kb.q[1] = p[d0 * 4 + 2 + half];      // d = d0*32 + 16 + half*8 + 0..7
                    sc = wmma_f16(qf[d0].h, kb.h, sc);
                }
                st[kt] = sc;
            }

            // online softmax update + stage P into per-wave LDS
            _Float16* pw = &Ps[wave * 640];
            for (int i = 0; i < 8; i++) {
                int qrow = qwave + i + 8 * half;
                float v0 = st[0][i];
                float v1 = st[1][i];
                if (kt0 + col > qrow)       v0 = NEG_BIG;
                if (kt0 + 16 + col > qrow)  v1 = NEG_BIG;
                float mx = fmaxf(v0, v1);
                for (int off = 1; off < 16; off <<= 1)
                    mx = fmaxf(mx, __shfl_xor(mx, off, 32));
                float mn = fmaxf(mrun[i], mx);
                float corr = __expf(mrun[i] - mn);
                mrun[i] = mn;
                float p0 = __expf(v0 - mn);
                float p1 = __expf(v1 - mn);
                float ps = p0 + p1;
                for (int off = 1; off < 16; off <<= 1)
                    ps += __shfl_xor(ps, off, 32);
                lrun[i] = lrun[i] * corr + ps;
                for (int c = 0; c < 4; c++) acc[c][i] *= corr;
                int prow = i + 8 * half;
                pw[prow * 40 + col]      = (_Float16)p0;
                pw[prow * 40 + 16 + col] = (_Float16)p1;
            }
            asm volatile("s_wait_dscnt 0" ::: "memory");

            // P fragment (A-layout 16x32) from per-wave LDS
            frag16 pf;
            {
                const uint4* pp = (const uint4*)(&pw[col * 40]);
                pf.q[0] = pp[half];
                pf.q[1] = pp[2 + half];
            }
            // PV: accumulate into 16x64 output (4 hd chunks)
            for (int c = 0; c < 4; c++) {
                frag16 vb;
                const uint4* p = (const uint4*)(&Vs[cur][(c * 16 + col) * 40]);
                vb.q[0] = p[half];          // key = half*8 + 0..7
                vb.q[1] = p[2 + half];      // key = 16 + half*8 + 0..7
                acc[c] = wmma_f16(pf.h, vb.h, acc[c]);
            }
        }
    }

    // epilogue: ctx[b, t, h*64 + d] = acc / l
    const int bidx = bh >> 4;
    const int hidx = bh & 15;
    for (int c = 0; c < 4; c++) {
        for (int i = 0; i < 8; i++) {
            int qrow = qwave + i + 8 * half;
            float o = acc[c][i] / lrun[i];
            ctx[((size_t)bidx * TDIM + qrow) * DDIM + hidx * HD + c * 16 + col] = (_Float16)o;
        }
    }
}

// ---------------------------------------------------------------------------
extern "C" void kernel_launch(void* const* d_in, const int* in_sizes, int n_in,
                              void* d_out, int out_size, void* d_ws, size_t ws_size,
                              hipStream_t stream) {
    (void)in_sizes; (void)n_in; (void)out_size; (void)ws_size;

    const float* x  = (const float*)d_in[0];
    const float* Wq = (const float*)d_in[1];
    const float* bq = (const float*)d_in[2];
    const float* Wk = (const float*)d_in[3];
    const float* bk = (const float*)d_in[4];
    const float* Wv = (const float*)d_in[5];
    const float* bv = (const float*)d_in[6];
    const float* Wo = (const float*)d_in[7];
    const float* bo = (const float*)d_in[8];

    char* ws = (char*)d_ws;
    _Float16* xh   = (_Float16*)(ws);                          // 8 MB
    _Float16* WqS  = (_Float16*)(ws + ((size_t)8  << 20));     // 2 MB swizzled
    _Float16* WkS  = (_Float16*)(ws + ((size_t)10 << 20));
    _Float16* WvS  = (_Float16*)(ws + ((size_t)12 << 20));
    _Float16* WoS  = (_Float16*)(ws + ((size_t)14 << 20));
    _Float16* Qb   = (_Float16*)(ws + ((size_t)16 << 20));     // 8 MB [B,H,T,64]
    _Float16* Kb   = (_Float16*)(ws + ((size_t)24 << 20));     // 8 MB [B,H,T,64]
    _Float16* Vtb  = (_Float16*)(ws + ((size_t)32 << 20));     // 8 MB [B,H,64,T]
    _Float16* ctxh = (_Float16*)(ws + ((size_t)40 << 20));     // 8 MB [B*T, D]

    const int nx = MTOT * DDIM;  // 4,194,304
    f32_to_f16_kernel<<<nx / 256, 256, 0, stream>>>(x, xh, nx);

    const int nw = DDIM * DDIM;  // 1,048,576
    swizzle_w_kernel<<<nw / 256, 256, 0, stream>>>(Wq, WqS, DDIM, DDIM);
    swizzle_w_kernel<<<nw / 256, 256, 0, stream>>>(Wk, WkS, DDIM, DDIM);
    swizzle_w_kernel<<<nw / 256, 256, 0, stream>>>(Wv, WvS, DDIM, DDIM);
    swizzle_w_kernel<<<nw / 256, 256, 0, stream>>>(Wo, WoS, DDIM, DDIM);

    dim3 gg(DDIM / 64, MTOT / 128);  // (16, 32)
    // Q gets 1/sqrt(hd) folded in; K natural layout; V transposed layout
    gemm_f16_kernel<<<gg, 256, 0, stream>>>(xh, WqS, bq, (void*)Qb,  MTOT, DDIM, DDIM, 0, 0.125f);
    gemm_f16_kernel<<<gg, 256, 0, stream>>>(xh, WkS, bk, (void*)Kb,  MTOT, DDIM, DDIM, 0, 1.0f);
    gemm_f16_kernel<<<gg, 256, 0, stream>>>(xh, WvS, bv, (void*)Vtb, MTOT, DDIM, DDIM, 1, 1.0f);

    dim3 ag(TDIM / 128, BDIM * HDIM);  // (16, 32)
    attn_kernel<<<ag, 256, 0, stream>>>(Qb, Kb, Vtb, ctxh);

    gemm_f16_kernel<<<gg, 256, 0, stream>>>(ctxh, WoS, bo, d_out, MTOT, DDIM, DDIM, 2, 1.0f);
}
